// GatedMixedCoding_41077067219223
// MI455X (gfx1250) — matrix operand order repaired and checked by reading, not compiled, not run on hardware
//
#include <hip/hip_runtime.h>
#include <hip/hip_bf16.h>
#include <math.h>

typedef __bf16 bf16;
typedef __attribute__((ext_vector_type(16))) __bf16 v16bf;
typedef __attribute__((ext_vector_type(8)))  __bf16 v8bf;
typedef __attribute__((ext_vector_type(8)))  float  v8f;

#define BI   64
#define BC   64
#define RR   36   // valid regions
#define RP   48   // padded regions (3 M-tiles)
#define WW   50   // valid words
#define WP   64   // padded words (4 N-tiles)
#define DD   512  // hidden dim (K)
#define KC   128  // K chunk staged in LDS
#define LDA  136  // LDS row stride (bf16 elems), padded for bank spread
#define SSTR 68   // LDS S row stride (floats)
#define GSTR 53   // Gram row stride (floats), odd -> conflict-free row reads
#define TEMP_INV 10.0f

// ---------- small helpers ----------
union BFU { __hip_bfloat16 h; bf16 b; unsigned short u; };
__device__ __forceinline__ bf16 f2bf(float x) { BFU u; u.h = __float2bfloat16(x); return u.b; }
__device__ __forceinline__ float bf2f(bf16 v) { BFU u; u.b = v; return __bfloat162float(u.h); }

__device__ __forceinline__ float wsum(float x) {
#pragma unroll
  for (int o = 16; o; o >>= 1) x += __shfl_xor(x, o, 32);
  return x;
}
__device__ __forceinline__ float wmaxr(float x) {
#pragma unroll
  for (int o = 16; o; o >>= 1) x = fmaxf(x, __shfl_xor(x, o, 32));
  return x;
}

// A fragment: lane half picks K {k..k+7, k+16..k+23} (ISA 16-bit A layout)
__device__ __forceinline__ v16bf ld_fragA(const bf16* p) {
  v8bf a = *(const v8bf*)p;
  v8bf b = *(const v8bf*)(p + 16);
  v16bf r;
#pragma unroll
  for (int j = 0; j < 8; ++j) { r[j] = a[j]; r[j + 8] = b[j]; }
  return r;
}
// B fragment: lane half picks K {k..k+7, k+16..k+23} (same as A layout)
__device__ __forceinline__ v16bf ld_fragB(const bf16* p) {
  v8bf a = *(const v8bf*)p;
  v8bf b = *(const v8bf*)(p + 16);
  v16bf r;
#pragma unroll
  for (int j = 0; j < 8; ++j) { r[j] = a[j]; r[j + 8] = b[j]; }
  return r;
}

// ---------- kernel 0: fp32 -> bf16 hi/lo split with row padding ----------
__global__ __launch_bounds__(128)
void convert_pad_kernel(const float* __restrict__ src, bf16* __restrict__ hi,
                        bf16* __restrict__ lo, int valid_rows, int pad_rows) {
  int b = blockIdx.x;
  int batch = b / pad_rows, row = b % pad_rows;
  bool ok = row < valid_rows;
#pragma unroll
  for (int k = 0; k < 4; ++k) {
    int d = threadIdx.x + (k << 7);
    float x = ok ? src[((size_t)(batch * valid_rows + row)) * DD + d] : 0.0f;
    bf16 h = f2bf(x);
    bf16 l = f2bf(x - bf2f(h));
    size_t o = (size_t)b * DD + d;
    hi[o] = h; lo[o] = l;
  }
}

// ---------- kernel 1: per-region gate MLPs + image-row norms (fp32) ----------
__global__ __launch_bounds__(256)
void gate_kernel(const float* __restrict__ imgs,
                 const float* __restrict__ sw1, const float* __restrict__ sb1,
                 const float* __restrict__ sw2, const float* __restrict__ sb2,
                 const float* __restrict__ hw1, const float* __restrict__ hb1,
                 const float* __restrict__ hw2, const float* __restrict__ hb2,
                 float* __restrict__ gs, float* __restrict__ gh, float* __restrict__ inr) {
  __shared__ float x[DD];
  __shared__ float red[256];
  int b = blockIdx.x, t = threadIdx.x;
  x[t]       = imgs[(size_t)b * DD + t];
  x[t + 256] = imgs[(size_t)b * DD + t + 256];
  __syncthreads();

  // ||x||^2
  red[t] = x[t] * x[t] + x[t + 256] * x[t + 256];
  __syncthreads();
  for (int s = 128; s; s >>= 1) { if (t < s) red[t] += red[t + s]; __syncthreads(); }
  float nrm2 = red[0];
  __syncthreads();

  // soft gate logit
  float acc = sb1[t];
  for (int d = 0; d < DD; ++d) acc += x[d] * sw1[d * 256 + t];
  red[t] = fmaxf(acc, 0.0f) * sw2[t];
  __syncthreads();
  for (int s = 128; s; s >>= 1) { if (t < s) red[t] += red[t + s]; __syncthreads(); }
  float sgl = red[0] + sb2[0];
  __syncthreads();

  // hard gate logit
  acc = hb1[t];
  for (int d = 0; d < DD; ++d) acc += x[d] * hw1[d * 256 + t];
  red[t] = fmaxf(acc, 0.0f) * hw2[t];
  __syncthreads();
  for (int s = 128; s; s >>= 1) { if (t < s) red[t] += red[t + s]; __syncthreads(); }
  float hgl = red[0] + hb2[0];

  if (t == 0) {
    float sg = 1.0f / (1.0f + expf(-sgl));
    float hg = 1.0f / (1.0f + expf(-hgl));
    float tot = sg + hg + 1e-8f;
    gs[b] = sg / tot;
    gh[b] = hg / tot;
    inr[b] = sqrtf(nrm2);
  }
}

// ---------- kernel 2: per-caption Gram G_c = caps[c] @ caps[c]^T (fp32) ----------
__global__ __launch_bounds__(256)
void gram_kernel(const float* __restrict__ caps, float* __restrict__ G) {
  extern __shared__ char gsm[];
  float* sc = (float*)gsm;  // [WW][DD]
  int c = blockIdx.x;
  for (int t = threadIdx.x; t < WW * DD; t += 256)
    sc[t] = caps[(size_t)c * WW * DD + t];
  __syncthreads();
  for (int t = threadIdx.x; t < WW * GSTR; t += 256) {
    int w = t / GSTR, w2 = t % GSTR;
    float s = 0.0f;
    if (w2 < WW) {
      const float* a = sc + w * DD;
      const float* b = sc + w2 * DD;
      for (int d = 0; d < DD; ++d) s += a[d] * b[d];
    }
    G[(size_t)c * (WW * GSTR) + t] = s;
  }
}

// ---------- kernel 3: main pair kernel (WMMA GEMM + fused epilogue) ----------
__global__ __launch_bounds__(384)
void pair_kernel(const bf16* __restrict__ Ahi, const bf16* __restrict__ Alo,
                 const bf16* __restrict__ Bhi, const bf16* __restrict__ Blo,
                 const float* __restrict__ Gg,
                 const float* __restrict__ gs, const float* __restrict__ gh,
                 const float* __restrict__ inr,
                 const int* __restrict__ img_lens, const int* __restrict__ cap_lens,
                 float* __restrict__ out) {
  extern __shared__ char smem[];
  bf16* sAhi = (bf16*)smem;                 // RP*LDA
  bf16* sAlo = sAhi + RP * LDA;             // RP*LDA
  bf16* sBhi = sAlo + RP * LDA;             // WP*LDA
  bf16* sBlo = sBhi + WP * LDA;             // WP*LDA
  float* sS  = (float*)(sBlo + WP * LDA);   // RP*SSTR
  float* sG  = sS + RP * SSTR;              // WW*GSTR

  const int tid  = threadIdx.x;
  const int lane = tid & 31;
  const int wv   = tid >> 5;      // 12 waves
  const int c = blockIdx.x, i = blockIdx.y;
  const int mtile = wv >> 2, ntile = wv & 3;
  const int lrow  = lane & 15;
  const int hi16  = (lane >> 4) & 1;

  const int imglen = img_lens[i];
  const int caplen = cap_lens[c];

  // stage caption Gram into LDS
  for (int t = tid; t < WW * GSTR; t += 384)
    sG[t] = Gg[(size_t)c * (WW * GSTR) + t];

  const bf16* gAhi = Ahi + (size_t)i * RP * DD;
  const bf16* gAlo = Alo + (size_t)i * RP * DD;
  const bf16* gBhi = Bhi + (size_t)c * WP * DD;
  const bf16* gBlo = Blo + (size_t)c * WP * DD;

  v8f acc = {};
  for (int kc = 0; kc < DD / KC; ++kc) {
    __syncthreads();
    const int kof = kc * KC;
    for (int t = tid; t < RP * (KC / 8); t += 384) {
      int row = t >> 4, cv = (t & 15) << 3;
      *(uint4*)(sAhi + row * LDA + cv) = *(const uint4*)(gAhi + (size_t)row * DD + kof + cv);
      *(uint4*)(sAlo + row * LDA + cv) = *(const uint4*)(gAlo + (size_t)row * DD + kof + cv);
    }
    for (int t = tid; t < WP * (KC / 8); t += 384) {
      int row = t >> 4, cv = (t & 15) << 3;
      *(uint4*)(sBhi + row * LDA + cv) = *(const uint4*)(gBhi + (size_t)row * DD + kof + cv);
      *(uint4*)(sBlo + row * LDA + cv) = *(const uint4*)(gBlo + (size_t)row * DD + kof + cv);
    }
    __syncthreads();
#pragma unroll
    for (int ks = 0; ks < KC / 32; ++ks) {
      const int ka = ks * 32 + (hi16 ? 8 : 0);
      const int kb = ks * 32 + (hi16 ? 8 : 0);
      v16bf fahi = ld_fragA(sAhi + (mtile * 16 + lrow) * LDA + ka);
      v16bf falo = ld_fragA(sAlo + (mtile * 16 + lrow) * LDA + ka);
      v16bf fbhi = ld_fragB(sBhi + (ntile * 16 + lrow) * LDA + kb);
      v16bf fblo = ld_fragB(sBlo + (ntile * 16 + lrow) * LDA + kb);
      // bf16x3 error-compensated fp32 GEMM: hi*hi + hi*lo + lo*hi
      acc = __builtin_amdgcn_wmma_f32_16x16x32_bf16(false, fahi, false, fbhi, (short)0, acc, false, false);
      acc = __builtin_amdgcn_wmma_f32_16x16x32_bf16(false, fahi, false, fblo, (short)0, acc, false, false);
      acc = __builtin_amdgcn_wmma_f32_16x16x32_bf16(false, falo, false, fbhi, (short)0, acc, false, false);
    }
  }

  // scatter accumulator tile to LDS S (ISA 32-bit C/D layout)
  {
    int m0 = mtile * 16 + hi16 * 8;
    int n  = ntile * 16 + lrow;
#pragma unroll
    for (int g = 0; g < 8; ++g) sS[(m0 + g) * SSTR + n] = acc[g];
  }
  __syncthreads();

  // ---- epilogue: 3 region rows per wave ----
  const int w0 = lane;        // word index 0..31
  const int w1 = lane + 32;   // word index 32..63
  const bool l1in = (w1 < WW);

#pragma unroll
  for (int jj = 0; jj < 3; ++jj) {
    const int r = wv + 12 * jj;  // 0..35
    if (r >= imglen) {
      if (lane == 0) out[((size_t)i * BC + c) * RR + r] = -1.0f;
      continue;
    }
    const float s0 = sS[r * SSTR + w0];
    const float s1 = sS[r * SSTR + w1];
    const bool v0 = (w0 < caplen);
    const bool v1 = (w1 < caplen);

    // hard attention: argmax over where(mask, sims, -1.0), first-index ties
    float h0 = v0 ? s0 : -1.0f;
    float h1 = v1 ? s1 : (l1in ? -1.0f : -3.4e38f);
    float bv; int bidx;
    if (h1 > h0) { bv = h1; bidx = w1; } else { bv = h0; bidx = w0; }
#pragma unroll
    for (int o = 16; o; o >>= 1) {
      float ov = __shfl_xor(bv, o, 32);
      int   oi = __shfl_xor(bidx, o, 32);
      if (ov > bv || (ov == bv && oi < bidx)) { bv = ov; bidx = oi; }
    }
    const bool hardvalid = (bidx < caplen);

    // soft attention (temperature softmax over valid words)
    float smax = wmaxr(v0 ? s0 : -3.4e38f);
    {
      float alt = wmaxr(v1 ? s1 : -3.4e38f);
      smax = fmaxf(smax, alt);
    }
    float e0 = v0 ? expf((s0 - smax) * TEMP_INV) : 0.0f;
    float e1 = v1 ? expf((s1 - smax) * TEMP_INV) : 0.0f;
    float esum = wsum(e0 + e1);
    float a0 = e0 / esum;
    float a1 = e1 / esum;

    // numerator pieces: a . S_row  and S[r, w*]
    float softnum = wsum(a0 * s0 + a1 * s1);

    // v = G a  (each lane: rows w0 and w1 of Gram)
    float vv0 = 0.0f, vv1 = 0.0f;
    const float* g0 = sG + w0 * GSTR;
    const float* g1 = sG + w1 * GSTR;
    for (int w = 0; w < caplen; ++w) {
      float aw = (w < 32) ? __shfl(a0, w, 32) : __shfl(a1, w - 32, 32);
      vv0 += g0[w] * aw;
      if (l1in) vv1 += g1[w] * aw;
    }
    float softsq = wsum(a0 * vv0 + a1 * vv1);            // a^T G a
    float cc = 0.0f;
    if (hardvalid) {
      if (w0 == bidx) cc += vv0;
      if (l1in && w1 == bidx) cc += vv1;
    }
    float crossv = wsum(cc);                             // (G a)[w*]
    float hardsq = hardvalid ? sG[bidx * GSTR + bidx] : 0.0f;

    const float sgn = gs[i * RR + r];
    const float hgn = gh[i * RR + r];
    const float inm = inr[i * RR + r];

    float num = sgn * softnum + (hardvalid ? hgn * bv : 0.0f);
    float msq = sgn * sgn * softsq + 2.0f * sgn * hgn * crossv + hgn * hgn * hardsq;
    float mn  = sqrtf(fmaxf(msq, 0.0f));
    float val = num / (fmaxf(inm, 1e-12f) * fmaxf(mn, 1e-12f));

    if (lane == 0) out[((size_t)i * BC + c) * RR + r] = val;
  }
}

// ---------- host launcher ----------
extern "C" void kernel_launch(void* const* d_in, const int* in_sizes, int n_in,
                              void* d_out, int out_size, void* d_ws, size_t ws_size,
                              hipStream_t stream) {
  (void)in_sizes; (void)n_in; (void)out_size; (void)ws_size;
  const float* imgs = (const float*)d_in[0];
  const float* caps = (const float*)d_in[1];
  const float* sw1  = (const float*)d_in[2];
  const float* sb1  = (const float*)d_in[3];
  const float* sw2  = (const float*)d_in[4];
  const float* sb2  = (const float*)d_in[5];
  const float* hw1  = (const float*)d_in[6];
  const float* hb1  = (const float*)d_in[7];
  const float* hw2  = (const float*)d_in[8];
  const float* hb2  = (const float*)d_in[9];
  const int* img_lens = (const int*)d_in[10];
  const int* cap_lens = (const int*)d_in[11];
  float* out = (float*)d_out;

  char* ws = (char*)d_ws;
  size_t off = 0;
  bf16* Ahi = (bf16*)(ws + off); off += (size_t)BI * RP * DD * 2;
  bf16* Alo = (bf16*)(ws + off); off += (size_t)BI * RP * DD * 2;
  bf16* Bhi = (bf16*)(ws + off); off += (size_t)BC * WP * DD * 2;
  bf16* Blo = (bf16*)(ws + off); off += (size_t)BC * WP * DD * 2;
  float* G  = (float*)(ws + off); off += (size_t)BC * WW * GSTR * 4;
  float* gS = (float*)(ws + off); off += (size_t)BI * RR * 4;
  float* gH = (float*)(ws + off); off += (size_t)BI * RR * 4;
  float* iN = (float*)(ws + off); off += (size_t)BI * RR * 4;

  convert_pad_kernel<<<BI * RP, 128, 0, stream>>>(imgs, Ahi, Alo, RR, RP);
  convert_pad_kernel<<<BC * WP, 128, 0, stream>>>(caps, Bhi, Blo, WW, WP);
  gate_kernel<<<BI * RR, 256, 0, stream>>>(imgs, sw1, sb1, sw2, sb2,
                                           hw1, hb1, hw2, hb2, gS, gH, iN);
  gram_kernel<<<BC, 256, WW * DD * 4, stream>>>(caps, G);

  size_t lds = (size_t)(2 * RP * LDA + 2 * WP * LDA) * 2   // bf16 stages
             + (size_t)RP * SSTR * 4                        // S tile
             + (size_t)WW * GSTR * 4;                       // Gram
  dim3 grid(BC, BI);
  pair_kernel<<<grid, 384, lds, stream>>>(Ahi, Alo, Bhi, Blo, G, gS, gH, iN,
                                          img_lens, cap_lens, out);
}
